// GroupQueryAttentionKVCache_38697655337183
// MI455X (gfx1250) — matrix operand order repaired and checked by reading, not compiled, not run on hardware
//
#include <hip/hip_runtime.h>
#include <math.h>

// ---------------- problem constants (fixed by reference) ----------------
#define D_MODEL 1024
#define N_GROUPS 4
#define GROUP_SIZE 4
#define D_H 64
#define CTX 512
#define Bsz 4
#define Ssz 2048
#define KV_N (N_GROUPS * D_H)   // 256

typedef __attribute__((ext_vector_type(16))) _Float16 v16h;
typedef __attribute__((ext_vector_type(8)))  float    v8f;

union FragH { v16h v; uint4 u4[2]; };
union AccF  { v8f  v; float f[8]; };

enum { EPI_F16 = 0, EPI_VT = 2, EPI_F32 = 3 };

__device__ __forceinline__ v8f wmma_f16(const v16h a, const v16h b, const v8f c) {
  // 8 args: (neg_a, A, neg_b, B, c_mod, C, reuse_a, reuse_b)
  return __builtin_amdgcn_wmma_f32_16x16x32_f16(false, a, false, b, (short)0, c,
                                                false, false);
}

// gfx1250 async VMEM->LDS copy, 16B per lane, tracked by ASYNCcnt.
__device__ __forceinline__ void async_b128(unsigned lds_off, const _Float16* gaddr) {
  asm volatile("global_load_async_to_lds_b128 %0, %1, off"
               :: "v"(lds_off), "v"(gaddr) : "memory");
}
__device__ __forceinline__ void wait_async0() {
  asm volatile("s_wait_asynccnt 0x0" ::: "memory");
}

// ---------------- conversion / transpose ----------------
__global__ void k_f32_to_f16(const float* __restrict__ in,
                             _Float16* __restrict__ out, int n) {
  int i = blockIdx.x * blockDim.x + threadIdx.x;
  if (i < n) out[i] = (_Float16)in[i];
}

// W[K][N] fp32 -> Wt[N][K] f16 (so B-fragments load row-contiguous)
__global__ void k_transpose_f16(const float* __restrict__ W,
                                _Float16* __restrict__ Wt, int K, int N) {
  int i = blockIdx.x * blockDim.x + threadIdx.x;
  if (i >= K * N) return;
  int k = i / N, n = i - k * N;
  Wt[(size_t)n * K + k] = (_Float16)W[i];
}

// ---------------- LDS-staged WMMA GEMM: C[M,N] = A[M,K] * Bt[N,K]^T + bias -------------
// 256 threads = 8 waves. Block tile 128(M) x 128(N); wave tile 32 x 64; K-step 32.
// Double-buffered LDS filled by async VMEM->LDS copies (no VGPR round-trip).
// LDS row stride padded to 40 halves (80B: 16B-aligned, banks skewed).
#define BK 32
#define LDA 40
__global__ __launch_bounds__(256)
void k_gemm_f16(const _Float16* __restrict__ A, const _Float16* __restrict__ Bt,
                const float* __restrict__ bias, void* __restrict__ out,
                int M, int N, int K, int mode, float scale) {
  __shared__ _Float16 As[2][128 * LDA];
  __shared__ _Float16 Bs[2][128 * LDA];
  const int tid = threadIdx.x;
  const int lane = tid & 31, wave = tid >> 5;
  const int r16 = lane & 15, half = lane >> 4;
  const int mi = wave & 3, ni = wave >> 2;            // 4 x 2 wave grid
  const int rBase = blockIdx.y * 128 + mi * 32;
  const int cBase = blockIdx.x * 128 + ni * 64;

  AccF acc[2][4];
#pragma unroll
  for (int u = 0; u < 2; ++u)
#pragma unroll
    for (int t = 0; t < 4; ++t)
#pragma unroll
      for (int r = 0; r < 8; ++r) acc[u][t].f[r] = 0.f;

  // staging: each thread moves 32B of A and 32B of B per K-step (2 async b128 each)
  const int srow = tid >> 1;                 // 0..127
  const int scol = (tid & 1) * 16;           // 0 or 16 halves
  const size_t agrow = (size_t)(blockIdx.y * 128 + srow) * K + scol;
  const size_t bgrow = (size_t)(blockIdx.x * 128 + srow) * K + scol;
  unsigned ldsA[2], ldsB[2];
#pragma unroll
  for (int p = 0; p < 2; ++p) {
    ldsA[p] = (unsigned)(uintptr_t)(&As[p][srow * LDA + scol]);
    ldsB[p] = (unsigned)(uintptr_t)(&Bs[p][srow * LDA + scol]);
  }

  // prologue: stage tile 0 into buffer 0
  async_b128(ldsA[0],      A + agrow);
  async_b128(ldsA[0] + 16, A + agrow + 8);
  async_b128(ldsB[0],      Bt + bgrow);
  async_b128(ldsB[0] + 16, Bt + bgrow + 8);
  wait_async0();
  __syncthreads();

  for (int k0 = 0; k0 < K; k0 += BK) {
    const int p = (k0 / BK) & 1;
    if (k0 + BK < K) {  // stage next tile into the other buffer while computing
      const int kn = k0 + BK;
      async_b128(ldsA[p ^ 1],      A + agrow + kn);
      async_b128(ldsA[p ^ 1] + 16, A + agrow + kn + 8);
      async_b128(ldsB[p ^ 1],      Bt + bgrow + kn);
      async_b128(ldsB[p ^ 1] + 16, Bt + bgrow + kn + 8);
    }

    // A 16x32 f16 layout: lanes0-15 K=[0..7]|[16..23], lanes16-31 K=[8..15]|[24..31]
    FragH a[2];
#pragma unroll
    for (int u = 0; u < 2; ++u) {
      const int ar = mi * 32 + u * 16 + r16;
      a[u].u4[0] = *(const uint4*)(&As[p][ar * LDA + half * 8]);
      a[u].u4[1] = *(const uint4*)(&As[p][ar * LDA + 16 + half * 8]);
    }
#pragma unroll
    for (int t = 0; t < 4; ++t) {
      // B 32x16 f16 layout: lane holds one column, K = half*8 + [0..7] | 16 + half*8 + [0..7]
      FragH b;
      const int br = ni * 64 + t * 16 + r16;
      b.u4[0] = *(const uint4*)(&Bs[p][br * LDA + half * 8]);
      b.u4[1] = *(const uint4*)(&Bs[p][br * LDA + 16 + half * 8]);
      acc[0][t].v = wmma_f16(a[0].v, b.v, acc[0][t].v);
      acc[1][t].v = wmma_f16(a[1].v, b.v, acc[1][t].v);
    }

    wait_async0();       // my async copies into p^1 have landed in LDS
    __syncthreads();     // everyone done computing p and copying p^1
  }

#pragma unroll
  for (int u = 0; u < 2; ++u) {
#pragma unroll
    for (int t = 0; t < 4; ++t) {
#pragma unroll
      for (int r = 0; r < 8; ++r) {
        const int row = rBase + u * 16 + r + 8 * half;  // C: M = vgpr + 8*(lane>=16)
        const int col = cBase + t * 16 + r16;           //    N = lane & 15
        const float v = (acc[u][t].f[r] + bias[col]) * scale;
        if (mode == EPI_F16) {
          ((_Float16*)out)[(size_t)row * N + col] = (_Float16)v;
        } else if (mode == EPI_VT) {
          // V epilogue: row = b*S+s, col = g*64+d -> vt[(b*256+col)*S + s]
          const int b = row >> 11, s = row & (Ssz - 1);
          ((_Float16*)out)[((size_t)(b * 256 + col)) * Ssz + s] = (_Float16)v;
        } else {
          ((float*)out)[(size_t)row * N + col] = v;
        }
      }
    }
  }
}

// ---------------- flash attention: block = (b, g, qblock), 4 waves = 4 heads ----------
// K/V tiles for a 32-key step staged once in LDS and shared by all heads of the group.
#define LDK 72    // 32 keys x 64 d, padded (144B rows, 16B aligned)
#define LDV 40    // 64 d x 32 keys, padded
#define LDP 40    // 16 q x 32 keys, padded
__global__ __launch_bounds__(128)
void k_attn(const _Float16* __restrict__ q, const _Float16* __restrict__ kmat,
            const _Float16* __restrict__ vt, _Float16* __restrict__ o) {
  __shared__ _Float16 Ks[32 * LDK];
  __shared__ _Float16 Vs[64 * LDV];
  __shared__ _Float16 Ps[GROUP_SIZE][16 * LDP];
  const int tid = threadIdx.x;
  const int lane = tid & 31;
  const int h = tid >> 5;                          // wave = head in group
  const int r16 = lane & 15, half = lane >> 4;
  const int bid = blockIdx.x;
  const int qb = bid & 127, g = (bid >> 7) & 3, b = bid >> 9;
  const int qbase = qb * 16;

  // Q fragments for d = 0..31 and 32..63 (1/sqrt(64) folded in at projection)
  FragH A0, A1;
  const size_t qrow = ((size_t)(b * Ssz + qbase + r16)) * D_MODEL + g * 256 + h * 64;
  A0.u4[0] = *(const uint4*)(q + qrow + half * 8);
  A0.u4[1] = *(const uint4*)(q + qrow + 16 + half * 8);
  A1.u4[0] = *(const uint4*)(q + qrow + 32 + half * 8);
  A1.u4[1] = *(const uint4*)(q + qrow + 48 + half * 8);

  float m8[8], l8[8];
  AccF O[4];
#pragma unroll
  for (int r = 0; r < 8; ++r) { m8[r] = -INFINITY; l8[r] = 0.f; }
#pragma unroll
  for (int t = 0; t < 4; ++t)
#pragma unroll
    for (int r = 0; r < 8; ++r) O[t].f[r] = 0.f;

  int jlo = qbase - (CTX - 1); if (jlo < 0) jlo = 0;
  const int jstart = ((jlo >> 4) & ~1) << 4;   // even 16-block; pre-block fully masked

  // staging coordinates (same for every iteration)
  const int skey = tid >> 2;                   // 0..31  (K rows)
  const int sdc  = (tid & 3) * 16;             // 0/16/32/48 within d
  const int sdr  = tid >> 1;                   // 0..63  (V rows = d)
  const int skc  = (tid & 1) * 16;             // 0/16 within keys

  for (int j = jstart; j <= qbase + 15; j += 32) {
    // ---- cooperative staging of K[32 keys][64 d] and V[64 d][32 keys] ----
    int keyg = j + skey; if (keyg > Ssz - 1) keyg = Ssz - 1;
    const size_t krow = ((size_t)(b * Ssz + keyg)) * KV_N + g * 64 + sdc;
    const uint4 kv0 = *(const uint4*)(kmat + krow);
    const uint4 kv1 = *(const uint4*)(kmat + krow + 8);
    const size_t vrow = ((size_t)(b * 256 + g * 64 + sdr)) * Ssz + j + skc;
    const uint4 vv0 = *(const uint4*)(vt + vrow);
    const uint4 vv1 = *(const uint4*)(vt + vrow + 8);
    __syncthreads();                            // previous iteration readers done
    *(uint4*)(Ks + skey * LDK + sdc)     = kv0;
    *(uint4*)(Ks + skey * LDK + sdc + 8) = kv1;
    *(uint4*)(Vs + sdr * LDV + skc)      = vv0;
    *(uint4*)(Vs + sdr * LDV + skc + 8)  = vv1;
    __syncthreads();

    // ---- scores: two 16x16 tiles, d accumulated in two k=32 WMMA steps ----
    AccF sc[2];
#pragma unroll
    for (int t = 0; t < 2; ++t) {
#pragma unroll
      for (int r = 0; r < 8; ++r) sc[t].f[r] = 0.f;
      FragH B0, B1;
      const int kr = t * 16 + r16;
      B0.u4[0] = *(const uint4*)(Ks + kr * LDK + half * 8);
      B0.u4[1] = *(const uint4*)(Ks + kr * LDK + 16 + half * 8);
      B1.u4[0] = *(const uint4*)(Ks + kr * LDK + 32 + half * 8);
      B1.u4[1] = *(const uint4*)(Ks + kr * LDK + 48 + half * 8);
      sc[t].v = wmma_f16(A0.v, B0.v, sc[t].v);
      sc[t].v = wmma_f16(A1.v, B1.v, sc[t].v);
    }

    // ---- mask + online softmax; row reductions over 16-lane halves ----
#pragma unroll
    for (int r = 0; r < 8; ++r) {
      const int row = r + 8 * half;
      const int i = qbase + row;
      float s0 = sc[0].f[r], s1 = sc[1].f[r];
      const int j0 = j + r16, j1 = j + 16 + r16;
      if (!((j0 <= i) && (j0 > i - CTX))) s0 = -INFINITY;
      if (!((j1 <= i) && (j1 > i - CTX))) s1 = -INFINITY;
      float mx = fmaxf(s0, s1);
      mx = fmaxf(mx, __shfl_xor(mx, 1, 32));
      mx = fmaxf(mx, __shfl_xor(mx, 2, 32));
      mx = fmaxf(mx, __shfl_xor(mx, 4, 32));
      mx = fmaxf(mx, __shfl_xor(mx, 8, 32));
      const float mnew = fmaxf(m8[r], mx);
      const float msub = (mnew == -INFINITY) ? 0.f : mnew;
      const float alpha = (m8[r] == -INFINITY) ? 0.f : __expf(m8[r] - msub);
      const float p0 = __expf(s0 - msub);
      const float p1 = __expf(s1 - msub);
      float ps = p0 + p1;
      ps += __shfl_xor(ps, 1, 32);
      ps += __shfl_xor(ps, 2, 32);
      ps += __shfl_xor(ps, 4, 32);
      ps += __shfl_xor(ps, 8, 32);
      l8[r] = l8[r] * alpha + ps;
      m8[r] = mnew;
#pragma unroll
      for (int t = 0; t < 4; ++t) O[t].f[r] *= alpha;
      Ps[h][row * LDP + r16]      = (_Float16)p0;   // per-wave P staging
      Ps[h][row * LDP + 16 + r16] = (_Float16)p1;
    }
    asm volatile("s_wait_dscnt 0x0" ::: "memory");  // intra-wave DS ordering for Ps

    // ---- P as A-fragment (16x32), V as B-fragments from shared LDS ----
    FragH P;
    P.u4[0] = *(const uint4*)(&Ps[h][r16 * LDP + half * 8]);
    P.u4[1] = *(const uint4*)(&Ps[h][r16 * LDP + 16 + half * 8]);
#pragma unroll
    for (int t = 0; t < 4; ++t) {
      FragH Vb;
      const int vr = t * 16 + r16;
      Vb.u4[0] = *(const uint4*)(Vs + vr * LDV + half * 8);
      Vb.u4[1] = *(const uint4*)(Vs + vr * LDV + 16 + half * 8);
      O[t].v = wmma_f16(P.v, Vb.v, O[t].v);
    }
  }

  // ---- normalize + store merged-head f16 output [B*S, D_MODEL] ----
#pragma unroll
  for (int r = 0; r < 8; ++r) {
    const int row = r + 8 * half;
    const float inv = 1.0f / l8[r];
#pragma unroll
    for (int t = 0; t < 4; ++t) {
      const size_t idx = ((size_t)(b * Ssz + qbase + row)) * D_MODEL +
                         g * 256 + h * 64 + t * 16 + r16;
      o[idx] = (_Float16)(O[t].f[r] * inv);
    }
  }
}

// ---------------- host orchestration ----------------
extern "C" void kernel_launch(void* const* d_in, const int* in_sizes, int n_in,
                              void* d_out, int out_size, void* d_ws, size_t ws_size,
                              hipStream_t stream) {
  const float* x  = (const float*)d_in[0];
  const float* Wq = (const float*)d_in[1];
  const float* bq = (const float*)d_in[2];
  const float* Wk = (const float*)d_in[3];
  const float* bk = (const float*)d_in[4];
  const float* Wv = (const float*)d_in[5];
  const float* bv = (const float*)d_in[6];
  const float* Wo = (const float*)d_in[7];
  const float* bo = (const float*)d_in[8];
  float* out = (float*)d_out;

  const int M = Bsz * Ssz;  // 8192

  // workspace carve-up (all f16): ~61 MB total
  _Float16* ws  = (_Float16*)d_ws;
  _Float16* xh  = ws; ws += (size_t)M * D_MODEL;
  _Float16* WqT = ws; ws += (size_t)D_MODEL * D_MODEL;
  _Float16* WkT = ws; ws += (size_t)KV_N * D_MODEL;
  _Float16* WvT = ws; ws += (size_t)KV_N * D_MODEL;
  _Float16* WoT = ws; ws += (size_t)D_MODEL * D_MODEL;
  _Float16* qh  = ws; ws += (size_t)M * D_MODEL;
  _Float16* kh  = ws; ws += (size_t)M * KV_N;
  _Float16* vth = ws; ws += (size_t)M * KV_N;
  _Float16* oh  = ws;

  const int nx = M * D_MODEL;
  k_f32_to_f16<<<(nx + 255) / 256, 256, 0, stream>>>(x, xh, nx);
  k_transpose_f16<<<(D_MODEL * D_MODEL + 255) / 256, 256, 0, stream>>>(Wq, WqT, D_MODEL, D_MODEL);
  k_transpose_f16<<<(D_MODEL * KV_N + 255) / 256, 256, 0, stream>>>(Wk, WkT, D_MODEL, KV_N);
  k_transpose_f16<<<(D_MODEL * KV_N + 255) / 256, 256, 0, stream>>>(Wv, WvT, D_MODEL, KV_N);
  k_transpose_f16<<<(D_MODEL * D_MODEL + 255) / 256, 256, 0, stream>>>(Wo, WoT, D_MODEL, D_MODEL);

  const dim3 blk(256);
  // Q: fold 1/sqrt(d_h)=0.125 into (xWq + bq)
  k_gemm_f16<<<dim3(D_MODEL / 128, M / 128), blk, 0, stream>>>(
      xh, WqT, bq, qh, M, D_MODEL, D_MODEL, EPI_F16, 0.125f);
  k_gemm_f16<<<dim3(KV_N / 128, M / 128), blk, 0, stream>>>(
      xh, WkT, bk, kh, M, KV_N, D_MODEL, EPI_F16, 1.0f);
  k_gemm_f16<<<dim3(KV_N / 128, M / 128), blk, 0, stream>>>(
      xh, WvT, bv, vth, M, KV_N, D_MODEL, EPI_VT, 1.0f);

  k_attn<<<Bsz * N_GROUPS * (Ssz / 16), 128, 0, stream>>>(qh, kh, vth, oh);

  k_gemm_f16<<<dim3(D_MODEL / 128, M / 128), blk, 0, stream>>>(
      oh, WoT, bo, out, M, D_MODEL, D_MODEL, EPI_F32, 1.0f);
}
